// ShuraMoE_78975858638966
// MI455X (gfx1250) — matrix operand order repaired and checked by reading, not compiled, not run on hardware
//
#include <hip/hip_runtime.h>
#include <math.h>
#include <stdint.h>

// ---------------- types ----------------
typedef unsigned int  u32;
typedef unsigned short u16;
typedef __attribute__((ext_vector_type(16))) __bf16 bf16x16;
typedef __attribute__((ext_vector_type(8)))  float  f32x8;

union FragAB {
  bf16x16 v;
  uint4   q[2];
};

// ---------------- problem constants ----------------
#define NTOK 8192     // B*N
#define DDIM 1024
#define FDIM 4096
#define NEXP 8
#define BM   64
#define BN   128
#define BK   32
#define MAXTM 128     // ceil(8192/64) worst-case tiles per expert

__device__ __forceinline__ u16 f2bf(float f) {
  u32 u = __float_as_uint(f);
  u32 r = u + 0x7FFFu + ((u >> 16) & 1u);
  return (u16)(r >> 16);
}

__device__ __forceinline__ float gelu_tanh(float x) {
  float x3 = x * x * x;
  return 0.5f * x * (1.0f + tanhf(0.7978845608028654f * (x + 0.044715f * x3)));
}

// ---- CDNA5 async global->LDS copy (ASYNCcnt-tracked, no VGPR data path) ----
__device__ __forceinline__ void async_ld_b128(u16* ldsDst, const u16* gSrc) {
  u32 l = (u32)(uintptr_t)ldsDst;     // generic LDS ptr: addr[31:0] is the LDS offset
  asm volatile("global_load_async_to_lds_b128 %0, %1, off"
               :: "v"(l), "v"(gSrc) : "memory");
}
__device__ __forceinline__ void wait_async0() {
  asm volatile("s_wait_asynccnt 0x0" ::: "memory");
}

// ---------------- kernel 0: zero output + counters ----------------
__global__ void k_init(float* __restrict__ out, int* __restrict__ counts) {
  size_t n = (size_t)NTOK * DDIM;
  size_t stride = (size_t)gridDim.x * blockDim.x;
  for (size_t i = (size_t)blockIdx.x * blockDim.x + threadIdx.x; i < n; i += stride)
    out[i] = 0.0f;
  if (blockIdx.x == 0 && threadIdx.x < NEXP) counts[threadIdx.x] = 0;
}

// ---------------- kernel: fp32 -> bf16 bulk convert (layout preserved) ----------------
__global__ void k_conv(const float* __restrict__ src, u16* __restrict__ dst, size_t n) {
  size_t stride = (size_t)gridDim.x * blockDim.x;
  for (size_t i = (size_t)blockIdx.x * blockDim.x + threadIdx.x; i < n; i += stride)
    dst[i] = f2bf(src[i]);
}

// ---------------- kernel: fp32 (K x N) -> bf16 (N x K) tiled transpose-convert --------
__global__ void __launch_bounds__(256)
k_tconv(const float* __restrict__ src, u16* __restrict__ dst, int K, int N) {
  __shared__ u16 tile[64][40];          // 80 B/row: keeps uint4 rows 16B-aligned
  int e  = blockIdx.z;
  int k0 = blockIdx.x * 32;
  int n0 = blockIdx.y * 64;
  int t  = threadIdx.x;

  int k  = t >> 3;                      // 0..31
  int nc = (t & 7) * 8;                 // 0..56
  const float* s = src + ((size_t)e * K + k0 + k) * N + n0 + nc;
  float4 f0 = *(const float4*)(s);
  float4 f1 = *(const float4*)(s + 4);
  tile[nc + 0][k] = f2bf(f0.x);
  tile[nc + 1][k] = f2bf(f0.y);
  tile[nc + 2][k] = f2bf(f0.z);
  tile[nc + 3][k] = f2bf(f0.w);
  tile[nc + 4][k] = f2bf(f1.x);
  tile[nc + 5][k] = f2bf(f1.y);
  tile[nc + 6][k] = f2bf(f1.z);
  tile[nc + 7][k] = f2bf(f1.w);
  __syncthreads();

  int n  = t >> 2;                      // 0..63
  int kc = (t & 3) * 8;                 // 0..24
  uint4 v = *(const uint4*)(&tile[n][kc]);
  *(uint4*)(dst + ((size_t)e * N + n0 + n) * K + k0 + kc) = v;
}

// ---------------- kernel: gating (wave per token) ----------------
__global__ void k_gate(const float* __restrict__ x, const float* __restrict__ gw,
                       int* __restrict__ counts, int* __restrict__ topkI,
                       float* __restrict__ topkW) {
  int wv   = threadIdx.x >> 5;
  int lane = threadIdx.x & 31;
  int token = blockIdx.x * 8 + wv;
  if (token >= NTOK) return;

  float acc[NEXP];
#pragma unroll
  for (int e = 0; e < NEXP; ++e) acc[e] = 0.0f;

  const float* xr = x + (size_t)token * DDIM;
  for (int d = lane; d < DDIM; d += 32) {
    float xv = xr[d];
    const float4* g = (const float4*)(gw + (size_t)d * NEXP);
    float4 g0 = g[0], g1 = g[1];
    acc[0] += xv * g0.x; acc[1] += xv * g0.y; acc[2] += xv * g0.z; acc[3] += xv * g0.w;
    acc[4] += xv * g1.x; acc[5] += xv * g1.y; acc[6] += xv * g1.z; acc[7] += xv * g1.w;
  }
#pragma unroll
  for (int e = 0; e < NEXP; ++e)
    for (int off = 16; off; off >>= 1)
      acc[e] += __shfl_down(acc[e], off, 32);

  if (lane == 0) {
    int i1 = 0;
#pragma unroll
    for (int e = 1; e < NEXP; ++e) if (acc[e] > acc[i1]) i1 = e;
    int i2 = (i1 == 0) ? 1 : 0;
#pragma unroll
    for (int e = 0; e < NEXP; ++e) if (e != i1 && acc[e] > acc[i2]) i2 = e;
    float e2 = __expf(acc[i2] - acc[i1]);
    float w1 = 1.0f / (1.0f + e2);
    float w2 = e2 / (1.0f + e2);
    topkI[token * 2 + 0] = i1;  topkW[token * 2 + 0] = w1;
    topkI[token * 2 + 1] = i2;  topkW[token * 2 + 1] = w2;
    atomicAdd(&counts[i1], 1);
    atomicAdd(&counts[i2], 1);
  }
}

// ---------------- kernel: exclusive scan over 8 counts ----------------
__global__ void k_scan(const int* __restrict__ counts, int* __restrict__ offsets,
                       int* __restrict__ cursor) {
  if (blockIdx.x == 0 && threadIdx.x == 0) {
    int s = 0;
    for (int e = 0; e < NEXP; ++e) { offsets[e] = s; cursor[e] = s; s += counts[e]; }
  }
}

// ---------------- kernel: scatter tokens into per-expert lists ----------------
__global__ void k_scatter(const int* __restrict__ topkI, const float* __restrict__ topkW,
                          int* __restrict__ cursor, int* __restrict__ tokList,
                          float* __restrict__ slotW) {
  int t = blockIdx.x * blockDim.x + threadIdx.x;
  if (t >= NTOK) return;
#pragma unroll
  for (int k = 0; k < 2; ++k) {
    int e = topkI[t * 2 + k];
    int p = atomicAdd(&cursor[e], 1);
    tokList[p] = t;
    slotW[p]   = topkW[t * 2 + k];
  }
}

// ======================= WMMA tile GEMM core (64 x 128 x 32) =======================
// 8 waves in 2x4 grid; each wave owns a 32x32 macro-tile (2x2 WMMA frags).
// LDS double-buffered with ASYNC global->LDS staging: one barrier per K-step.

// ---------------- GEMM1: h = gelu(x_gathered @ w1[e]),  K=1024, N=4096 ----------------
__global__ void __launch_bounds__(256)
k_gemm1(const u16* __restrict__ xbf, const u16* __restrict__ w1t,
        const int* __restrict__ tokList, const int* __restrict__ counts,
        const int* __restrict__ offsets, u16* __restrict__ hbuf) {
  __shared__ u16 sA[2][BM * BK];   // [row][k]
  __shared__ u16 sB[2][BN * BK];   // [n][k]  (weights pre-transposed)
  __shared__ int sTok[BM];

  int e  = blockIdx.x / MAXTM;
  int tm = blockIdx.x % MAXTM;
  int cnt = counts[e];
  if (tm * BM >= cnt) return;
  int offE = offsets[e];
  int n0base = blockIdx.y * BN;

  int t = threadIdx.x;
  if (t < BM) {
    int r = tm * BM + t;
    sTok[t] = (r < cnt) ? tokList[offE + r] : -1;
  }
  __syncthreads();

  // ---- loop-invariant staging bases (branchless: clamp invalid rows to row 0) ----
  int ar = t >> 2, ac = (t & 3) * 8;              // A-fill: 64 rows x 4 chunks of 8
  int tok  = sTok[ar];
  int tokC = (tok >= 0) ? tok : sTok[0];          // sTok[0] always valid
  const u16* aSrc = xbf + (size_t)tokC * DDIM + ac;

  int bn0 = t >> 2, bkc0 = (t & 3) * 8;           // B chunk 0
  int bn1 = (t + 256) >> 2, bkc1 = bkc0;          // B chunk 1
  const u16* bSrc0 = w1t + ((size_t)e * FDIM + n0base + bn0) * DDIM + bkc0;
  const u16* bSrc1 = w1t + ((size_t)e * FDIM + n0base + bn1) * DDIM + bkc1;

  u16* aDst[2] = { &sA[0][ar * BK + ac],  &sA[1][ar * BK + ac] };
  u16* bDst0[2] = { &sB[0][bn0 * BK + bkc0], &sB[1][bn0 * BK + bkc0] };
  u16* bDst1[2] = { &sB[0][bn1 * BK + bkc1], &sB[1][bn1 * BK + bkc1] };

  int lane = t & 31, wv = t >> 5;
  int hi = lane >> 4, ln = lane & 15;
  int r0 = (wv & 1) * 32;
  int c0 = (wv >> 1) * 32;

  const u16* aFrag[2] = { &sA[0][(r0 + ln) * BK + hi * 8],
                          &sA[1][(r0 + ln) * BK + hi * 8] };
  const u16* bFrag[2] = { &sB[0][(c0 + ln) * BK + hi * 8],
                          &sB[1][(c0 + ln) * BK + hi * 8] };

  f32x8 acc[2][2];
#pragma unroll
  for (int i = 0; i < 2; ++i)
#pragma unroll
    for (int j = 0; j < 2; ++j)
      acc[i][j] = (f32x8){0.f,0.f,0.f,0.f,0.f,0.f,0.f,0.f};

  // ---- prologue: async-stage tile kb=0 into buffer 0 ----
  async_ld_b128(aDst[0],  aSrc);
  async_ld_b128(bDst0[0], bSrc0);
  async_ld_b128(bDst1[0], bSrc1);
  wait_async0();
  __syncthreads();

  int cur = 0;
  for (int kb = 0; kb < DDIM; kb += BK) {
    int nkb = kb + BK;
    if (nkb < DDIM) {                   // async-stage next tile into other buffer
      async_ld_b128(aDst[cur ^ 1],  aSrc + nkb);
      async_ld_b128(bDst0[cur ^ 1], bSrc0 + nkb);
      async_ld_b128(bDst1[cur ^ 1], bSrc1 + nkb);
    }

    FragAB fa[2], fb[2];
#pragma unroll
    for (int i = 0; i < 2; ++i) {
      const u16* ap = aFrag[cur] + i * 16 * BK;
      fa[i].q[0] = *(const uint4*)(ap);
      fa[i].q[1] = *(const uint4*)(ap + 16);
    }
#pragma unroll
    for (int j = 0; j < 2; ++j) {
      const u16* bp = bFrag[cur] + j * 16 * BK;
      fb[j].q[0] = *(const uint4*)(bp);
      fb[j].q[1] = *(const uint4*)(bp + 16);
    }
#pragma unroll
    for (int i = 0; i < 2; ++i)
#pragma unroll
      for (int j = 0; j < 2; ++j)
        acc[i][j] = __builtin_amdgcn_wmma_f32_16x16x32_bf16(
            false, fa[i].v, false, fb[j].v, (short)0, acc[i][j], false, false);

    wait_async0();                      // our async writes to buf^1 landed
    __syncthreads();                    // everyone's landed; reads of buf done
    cur ^= 1;
  }

  // ---- epilogue: gelu + store bf16 into compact h buffer ----
#pragma unroll
  for (int i = 0; i < 2; ++i) {
#pragma unroll
    for (int v = 0; v < 8; ++v) {
      int m = v + hi * 8;
      int r = tm * BM + r0 + i * 16 + m;
      if (r < cnt) {
        size_t rowo = (size_t)(offE + r) * FDIM + n0base + c0 + ln;
        hbuf[rowo]      = f2bf(gelu_tanh(acc[i][0][v]));
        hbuf[rowo + 16] = f2bf(gelu_tanh(acc[i][1][v]));
      }
    }
  }
}

// ---------------- GEMM2: out[token] += w * (h @ w2[e]),  K=4096, N=1024 ----------------
__global__ void __launch_bounds__(256)
k_gemm2(const u16* __restrict__ hbuf, const u16* __restrict__ w2t,
        const int* __restrict__ tokList, const float* __restrict__ slotW,
        const int* __restrict__ counts, const int* __restrict__ offsets,
        float* __restrict__ out) {
  __shared__ u16 sA[2][BM * BK];
  __shared__ u16 sB[2][BN * BK];
  __shared__ int   sTok[BM];
  __shared__ float sWgt[BM];

  int e  = blockIdx.x / MAXTM;
  int tm = blockIdx.x % MAXTM;
  int cnt = counts[e];
  if (tm * BM >= cnt) return;
  int offE = offsets[e];
  int n0base = blockIdx.y * BN;

  int t = threadIdx.x;
  if (t < BM) {
    int r = tm * BM + t;
    bool ok = (r < cnt);
    sTok[t] = ok ? tokList[offE + r] : -1;
    sWgt[t] = ok ? slotW[offE + r] : 0.0f;
  }
  __syncthreads();

  int ar = t >> 2, ac = (t & 3) * 8;
  int rA = tm * BM + ar;
  int rC = (rA < cnt) ? rA : (cnt - 1);           // clamp: invalid rows re-read last row
  const u16* aSrc = hbuf + (size_t)(offE + rC) * FDIM + ac;

  int bn0 = t >> 2, bkc0 = (t & 3) * 8;
  int bn1 = (t + 256) >> 2, bkc1 = bkc0;
  const u16* bSrc0 = w2t + ((size_t)e * DDIM + n0base + bn0) * FDIM + bkc0;
  const u16* bSrc1 = w2t + ((size_t)e * DDIM + n0base + bn1) * FDIM + bkc1;

  u16* aDst[2] = { &sA[0][ar * BK + ac],  &sA[1][ar * BK + ac] };
  u16* bDst0[2] = { &sB[0][bn0 * BK + bkc0], &sB[1][bn0 * BK + bkc0] };
  u16* bDst1[2] = { &sB[0][bn1 * BK + bkc1], &sB[1][bn1 * BK + bkc1] };

  int lane = t & 31, wv = t >> 5;
  int hi = lane >> 4, ln = lane & 15;
  int r0 = (wv & 1) * 32;
  int c0 = (wv >> 1) * 32;

  const u16* aFrag[2] = { &sA[0][(r0 + ln) * BK + hi * 8],
                          &sA[1][(r0 + ln) * BK + hi * 8] };
  const u16* bFrag[2] = { &sB[0][(c0 + ln) * BK + hi * 8],
                          &sB[1][(c0 + ln) * BK + hi * 8] };

  f32x8 acc[2][2];
#pragma unroll
  for (int i = 0; i < 2; ++i)
#pragma unroll
    for (int j = 0; j < 2; ++j)
      acc[i][j] = (f32x8){0.f,0.f,0.f,0.f,0.f,0.f,0.f,0.f};

  async_ld_b128(aDst[0],  aSrc);
  async_ld_b128(bDst0[0], bSrc0);
  async_ld_b128(bDst1[0], bSrc1);
  wait_async0();
  __syncthreads();

  int cur = 0;
  for (int kb = 0; kb < FDIM; kb += BK) {
    int nkb = kb + BK;
    if (nkb < FDIM) {
      async_ld_b128(aDst[cur ^ 1],  aSrc + nkb);
      async_ld_b128(bDst0[cur ^ 1], bSrc0 + nkb);
      async_ld_b128(bDst1[cur ^ 1], bSrc1 + nkb);
    }

    FragAB fa[2], fb[2];
#pragma unroll
    for (int i = 0; i < 2; ++i) {
      const u16* ap = aFrag[cur] + i * 16 * BK;
      fa[i].q[0] = *(const uint4*)(ap);
      fa[i].q[1] = *(const uint4*)(ap + 16);
    }
#pragma unroll
    for (int j = 0; j < 2; ++j) {
      const u16* bp = bFrag[cur] + j * 16 * BK;
      fb[j].q[0] = *(const uint4*)(bp);
      fb[j].q[1] = *(const uint4*)(bp + 16);
    }
#pragma unroll
    for (int i = 0; i < 2; ++i)
#pragma unroll
      for (int j = 0; j < 2; ++j)
        acc[i][j] = __builtin_amdgcn_wmma_f32_16x16x32_bf16(
            false, fa[i].v, false, fb[j].v, (short)0, acc[i][j], false, false);

    wait_async0();
    __syncthreads();
    cur ^= 1;
  }

#pragma unroll
  for (int i = 0; i < 2; ++i) {
#pragma unroll
    for (int v = 0; v < 8; ++v) {
      int m = v + hi * 8;
      int tr = r0 + i * 16 + m;
      int tok = sTok[tr];
      if (tok >= 0) {
        float wgt = sWgt[tr];
        size_t rowo = (size_t)tok * DDIM + n0base + c0 + ln;
        atomicAdd(&out[rowo],      wgt * acc[i][0][v]);
        atomicAdd(&out[rowo + 16], wgt * acc[i][1][v]);
      }
    }
  }
}

// ---------------- host launcher ----------------
extern "C" void kernel_launch(void* const* d_in, const int* in_sizes, int n_in,
                              void* d_out, int out_size, void* d_ws, size_t ws_size,
                              hipStream_t stream) {
  const float* x  = (const float*)d_in[0];   // (4,2048,1024)
  const float* gw = (const float*)d_in[1];   // (1024,8)
  const float* w1 = (const float*)d_in[2];   // (8,1024,4096)  K=D, N=F
  const float* w2 = (const float*)d_in[3];   // (8,4096,1024)  K=F, N=D
  float* out = (float*)d_out;

  char* ws = (char*)d_ws;
  size_t off = 0;
  int*   counts  = (int*)(ws + off);  off += 256;
  int*   offsets = (int*)(ws + off);  off += 256;
  int*   cursor  = (int*)(ws + off);  off += 256;
  int*   topkI   = (int*)(ws + off);  off += (size_t)NTOK * 2 * sizeof(int);
  float* topkW   = (float*)(ws + off); off += (size_t)NTOK * 2 * sizeof(float);
  int*   tokList = (int*)(ws + off);  off += (size_t)NTOK * 2 * sizeof(int);
  float* slotW   = (float*)(ws + off); off += (size_t)NTOK * 2 * sizeof(float);
  off = (off + 255) & ~(size_t)255;
  u16* xbf = (u16*)(ws + off); off += (size_t)NTOK * DDIM * sizeof(u16);
  u16* w1t = (u16*)(ws + off); off += (size_t)NEXP * DDIM * FDIM * sizeof(u16); // [e][F][D]
  u16* w2t = (u16*)(ws + off); off += (size_t)NEXP * FDIM * DDIM * sizeof(u16); // [e][D][F]
  u16* hbuf = (u16*)(ws + off); off += (size_t)NTOK * 2 * FDIM * sizeof(u16);

  k_init<<<2048, 256, 0, stream>>>(out, counts);
  k_tconv<<<dim3(DDIM / 32, FDIM / 64, NEXP), 256, 0, stream>>>(w1, w1t, DDIM, FDIM);
  k_tconv<<<dim3(FDIM / 32, DDIM / 64, NEXP), 256, 0, stream>>>(w2, w2t, FDIM, DDIM);
  k_conv<<<1024, 256, 0, stream>>>(x, xbf, (size_t)NTOK * DDIM);
  k_gate<<<NTOK / 8, 256, 0, stream>>>(x, gw, counts, topkI, topkW);
  k_scan<<<1, 32, 0, stream>>>(counts, offsets, cursor);
  k_scatter<<<NTOK / 256, 256, 0, stream>>>(topkI, topkW, cursor, tokList, slotW);
  k_gemm1<<<dim3(NEXP * MAXTM, FDIM / BN), 256, 0, stream>>>(
      xbf, w1t, tokList, counts, offsets, hbuf);
  k_gemm2<<<dim3(NEXP * MAXTM, DDIM / BN), 256, 0, stream>>>(
      hbuf, w2t, tokList, slotW, counts, offsets, out);
}
